// FlashSelfAttention_46832323396169
// MI455X (gfx1250) — matrix operand order repaired and checked by reading, not compiled, not run on hardware
//
#include <hip/hip_runtime.h>

// FlashSelfAttention fwd: B=2,S=2048,H=8,D=64 fp16, bias (B,H,S,S), gates (B,H,1,S).
// Memory-bound on the 134MB bias stream (roofline floor ~7us @ 23.3TB/s).
// Flash attention, double-buffered: tile t+1's bias/K stream into LDS via
// global_load_async_to_lds_b128 (bias marked non-temporal) while tile t runs
// through v_wmma_f32_16x16x32_f16 (QK^T and PV) + online softmax.

typedef __attribute__((ext_vector_type(16))) _Float16 v16h;
typedef __attribute__((ext_vector_type(8)))  _Float16 v8h;
typedef __attribute__((ext_vector_type(8)))  float    v8f;

#define B_ 2
#define S_ 2048
#define H_ 8
#define D_ 64
#define QKV_SSTRIDE (3 * H_ * D_)   // 1536 halves between consecutive s
#define KT 32                       // key-tile (WMMA K for fp16)
#define NTILES (S_ / KT)            // 64
#define WAVES 8
#define VT_STRIDE 40                // padded LDS row stride for transposed V

static __device__ __forceinline__ v16h mk16(v8h lo, v8h hi) {
  v16h r;
#pragma unroll
  for (int i = 0; i < 8; ++i) { r[i] = lo[i]; r[i + 8] = hi[i]; }
  return r;
}

static __device__ __forceinline__ v8f wmma16(v16h a, v16h b, v8f c) {
  return __builtin_amdgcn_wmma_f32_16x16x32_f16(false, a, false, b, (short)0, c,
                                                false, false);
}

// Async 16B copies global -> LDS (ASYNCcnt).  INST_OFFSET applies to both the
// global and LDS addresses.  Bias uses non-temporal hint (read-once stream).
static __device__ __forceinline__ void async_b128(unsigned lds_off,
                                                  unsigned long long ga) {
  asm volatile("global_load_async_to_lds_b128 %0, %1, off"
               :: "v"(lds_off), "v"(ga) : "memory");
}
static __device__ __forceinline__ void async_b128_nt(unsigned lds_off,
                                                     unsigned long long ga) {
  asm volatile("global_load_async_to_lds_b128 %0, %1, off th:TH_LOAD_NT"
               :: "v"(lds_off), "v"(ga) : "memory");
}
static __device__ __forceinline__ void async_b128_nt_o32(unsigned lds_off,
                                                         unsigned long long ga) {
  asm volatile("global_load_async_to_lds_b128 %0, %1, off offset:32 th:TH_LOAD_NT"
               :: "v"(lds_off), "v"(ga) : "memory");
}
static __device__ __forceinline__ void wait_async3() {   // older tile done
  asm volatile("s_wait_asynccnt 0x3" ::: "memory");
}
static __device__ __forceinline__ void wait_async0() {
  asm volatile("s_wait_asynccnt 0x0" ::: "memory");
}
static __device__ __forceinline__ void wait_dscnt0() {
  asm volatile("s_wait_dscnt 0x0" ::: "memory");
}

__global__ __launch_bounds__(256)
void fa_fwd_kernel(const _Float16* __restrict__ qkv,
                   const _Float16* __restrict__ bias,
                   const _Float16* __restrict__ gates,
                   _Float16* __restrict__ out) {
  __shared__ __align__(16) _Float16 ldsK[2][KT * D_];           // [k][d]
  __shared__ __align__(16) _Float16 ldsVt[2][D_ * VT_STRIDE];   // [d][k]
  __shared__ __align__(16) _Float16 ldsB[2][WAVES][16 * KT];    // bias, then P
  __shared__ __align__(16) _Float16 ldsG[S_];                   // gates row

  const int tid  = threadIdx.x;
  const int wave = tid >> 5;
  const int lane = tid & 31;

  const int qblk = blockIdx.x & 15;
  const int bh   = blockIdx.x >> 4;
  const int h    = bh & (H_ - 1);
  const int b    = bh >> 3;
  const int qrow0 = qblk * 128 + wave * 16;

  const _Float16* qBase = qkv + ((size_t)(b * S_) * 3 + 0) * (H_ * D_) + h * D_;
  const _Float16* kBase = qkv + ((size_t)(b * S_) * 3 + 1) * (H_ * D_) + h * D_;
  const _Float16* vBase = qkv + ((size_t)(b * S_) * 3 + 2) * (H_ * D_) + h * D_;
  const _Float16* biasBase = bias + ((size_t)(b * H_ + h) * S_ + qrow0) * S_;
  const _Float16* gateBase = gates + (size_t)(b * H_ + h) * S_;

  // staging decompositions
  const int sr = tid >> 3;            // K/V tile row 0..31
  const int sc = (tid & 7) * 8;       // K/V tile col, 8-half chunks
  const int brow = lane & 15;         // bias tile row
  const int bcol = (lane >> 4) * 8;   // bias tile col base

  const unsigned kOff[2] = {(unsigned)(size_t)&ldsK[0][sr * D_ + sc],
                            (unsigned)(size_t)&ldsK[1][sr * D_ + sc]};
  const unsigned bOff[2] = {(unsigned)(size_t)&ldsB[0][wave][brow * KT + bcol],
                            (unsigned)(size_t)&ldsB[1][wave][brow * KT + bcol]};

  auto issue_tile = [&](int t0, int buf) {
    async_b128(kOff[buf],
        (unsigned long long)(size_t)(kBase + (size_t)(t0 + sr) * QKV_SSTRIDE + sc));
    const unsigned long long ba =
        (unsigned long long)(size_t)(biasBase + (size_t)brow * S_ + t0 + bcol);
    async_b128_nt(bOff[buf], ba);
    async_b128_nt_o32(bOff[buf], ba);
  };

  // ---- prologue: start tile 0, stage gates, load Q fragments ----
  issue_tile(0, 0);
  *(v8h*)&ldsG[tid * 8] = *(const v8h*)(gateBase + tid * 8);
  v8h vv = *(const v8h*)(vBase + (size_t)sr * QKV_SSTRIDE + sc);

  const int arow = lane & 15;
  const int aoff = (lane >> 4) * 8;
  const _Float16* qrow = qBase + (size_t)(qrow0 + arow) * QKV_SSTRIDE;
  const v16h aq0 = mk16(*(const v8h*)(qrow + 0  + aoff),
                        *(const v8h*)(qrow + 16 + aoff));
  const v16h aq1 = mk16(*(const v8h*)(qrow + 32 + aoff),
                        *(const v8h*)(qrow + 48 + aoff));

  const int cm  = (lane >> 4) * 8;
  const int cn  = lane & 15;
  const int bks = (lane >> 4) * 8;

  v8f o0 = {}, o1 = {}, o2 = {}, o3 = {};
  float mrow[8], lrow[8];
#pragma unroll
  for (int r = 0; r < 8; ++r) { mrow[r] = -3.0e38f; lrow[r] = 0.0f; }

  for (int it = 0; it < NTILES; ++it) {
    const int t0  = it * KT;
    const int cur = it & 1;

    // drain own LDS reads (prev tile's P/bias/K fragments), then block-wide
    // rendezvous so double-buffered LDS can be rewritten.
    wait_dscnt0();
    __syncthreads();

    // scatter V tile 'it' (loaded last iteration) into transposed LDS
    {
      _Float16* vt = &ldsVt[cur][0];
#pragma unroll
      for (int i = 0; i < 8; ++i) vt[(sc + i) * VT_STRIDE + sr] = vv[i];
    }
    // kick off tile it+1 (runs during compute of tile it)
    if (it + 1 < NTILES) {
      issue_tile(t0 + KT, cur ^ 1);
      vv = *(const v8h*)(vBase + (size_t)(t0 + KT + sr) * QKV_SSTRIDE + sc);
      wait_async3();   // tile it's 3 asyncs done; it+1's remain in flight
    } else {
      wait_async0();
    }
    __syncthreads();

    // ---- S = (QK^T)*scale + bias, *gate ----
    const _Float16* kb = &ldsK[cur][0];
    auto ldK = [&](int col, int kchunk) -> v16h {
      const _Float16* p = kb + col * D_ + kchunk + bks;
      return mk16(*(const v8h*)p, *(const v8h*)(p + 16));
    };
    const v16h bk00 = ldK(cn, 0),      bk01 = ldK(cn, 32);
    const v16h bk10 = ldK(16 + cn, 0), bk11 = ldK(16 + cn, 32);

    v8f s0 = {}; s0 = wmma16(aq0, bk00, s0); s0 = wmma16(aq1, bk01, s0);
    v8f s1 = {}; s1 = wmma16(aq0, bk10, s1); s1 = wmma16(aq1, bk11, s1);

    const float g0 = (float)ldsG[t0 + cn];
    const float g1 = (float)ldsG[t0 + 16 + cn];

    _Float16* const Pw = &ldsB[cur][wave][0];
    float p0[8], p1[8];
#pragma unroll
    for (int r = 0; r < 8; ++r) {
      const int m = cm + r;
      const float b0 = (float)Pw[m * KT + cn];
      const float b1 = (float)Pw[m * KT + 16 + cn];
      float x0 = (s0[r] * 0.125f + b0) * g0;
      float x1 = (s1[r] * 0.125f + b1) * g1;

      float tmax = fmaxf(x0, x1);
      tmax = fmaxf(tmax, __shfl_xor(tmax, 1, 32));
      tmax = fmaxf(tmax, __shfl_xor(tmax, 2, 32));
      tmax = fmaxf(tmax, __shfl_xor(tmax, 4, 32));
      tmax = fmaxf(tmax, __shfl_xor(tmax, 8, 32));

      const float mnew = fmaxf(mrow[r], tmax);
      const float corr = __expf(mrow[r] - mnew);
      mrow[r] = mnew;
      x0 = __expf(x0 - mnew);
      x1 = __expf(x1 - mnew);

      float rs = x0 + x1;
      rs += __shfl_xor(rs, 1, 32);
      rs += __shfl_xor(rs, 2, 32);
      rs += __shfl_xor(rs, 4, 32);
      rs += __shfl_xor(rs, 8, 32);

      lrow[r] = lrow[r] * corr + rs;
      o0[r] *= corr; o1[r] *= corr; o2[r] *= corr; o3[r] *= corr;
      p0[r] = x0; p1[r] = x1;
    }

    // C-layout -> A-layout through per-wave LDS (bias fully consumed; the
    // wave's LDS ops are in-order so no barrier needed)
#pragma unroll
    for (int r = 0; r < 8; ++r) {
      const int m = cm + r;
      Pw[m * KT + cn]      = (_Float16)p0[r];
      Pw[m * KT + 16 + cn] = (_Float16)p1[r];
    }
    const v16h pa = mk16(*(const v8h*)&Pw[arow * KT + aoff],
                         *(const v8h*)&Pw[arow * KT + 16 + aoff]);

    // ---- O += P x V ----
    const _Float16* vt = &ldsVt[cur][0];
    auto ldV = [&](int nb) -> v16h {
      const _Float16* p = vt + (nb * 16 + cn) * VT_STRIDE + bks;
      return mk16(*(const v8h*)p, *(const v8h*)(p + 16));
    };
    o0 = wmma16(pa, ldV(0), o0);
    o1 = wmma16(pa, ldV(1), o1);
    o2 = wmma16(pa, ldV(2), o2);
    o3 = wmma16(pa, ldV(3), o3);
  }

  // ---- epilogue: normalize, non-temporal fp16 stores ----
#pragma unroll
  for (int r = 0; r < 8; ++r) {
    const float inv = 1.0f / lrow[r];
    _Float16* orow = out + ((size_t)(b * S_ + qrow0 + cm + r) * H_ + h) * D_;
    __builtin_nontemporal_store((_Float16)(o0[r] * inv), &orow[0 * 16 + cn]);
    __builtin_nontemporal_store((_Float16)(o1[r] * inv), &orow[1 * 16 + cn]);
    __builtin_nontemporal_store((_Float16)(o2[r] * inv), &orow[2 * 16 + cn]);
    __builtin_nontemporal_store((_Float16)(o3[r] * inv), &orow[3 * 16 + cn]);
  }
}

extern "C" void kernel_launch(void* const* d_in, const int* in_sizes, int n_in,
                              void* d_out, int out_size, void* d_ws, size_t ws_size,
                              hipStream_t stream) {
  (void)in_sizes; (void)n_in; (void)out_size; (void)d_ws; (void)ws_size;
  const _Float16* qkv   = (const _Float16*)d_in[0];
  const _Float16* bias  = (const _Float16*)d_in[1];
  const _Float16* gates = (const _Float16*)d_in[2];
  _Float16* out = (_Float16*)d_out;

  dim3 grid(B_ * H_ * (S_ / 128));  // 256 blocks
  dim3 block(256);                  // 8 waves, one 16-row Q tile each
  fa_fwd_kernel<<<grid, block, 0, stream>>>(qkv, bias, gates, out);
}
